// DynAAMPG_90580860273217
// MI455X (gfx1250) — matrix operand order repaired and checked by reading, not compiled, not run on hardware
//
#include <hip/hip_runtime.h>

#define NN    20000   // nodes
#define NE    160000  // edges
#define DIN   128
#define HID   64
#define NH    8
#define HHID  512     // NH*HID
#define NGR   64
#define NCL   10

typedef __attribute__((ext_vector_type(16))) __bf16 v16bf;
typedef __attribute__((ext_vector_type(8)))  __bf16 v8bf;
typedef __attribute__((ext_vector_type(8)))  float  v8f;

__device__ inline unsigned short f2bf(float f) {
  unsigned int x = __float_as_uint(f);
  unsigned int r = x + 0x7fffu + ((x >> 16) & 1u);   // round-to-nearest-even
  return (unsigned short)(r >> 16);
}
__device__ inline float bf2f(unsigned short u) {
  return __uint_as_float(((unsigned int)u) << 16);
}

// ---------------------------------------------------------------- utilities
__global__ void fill_f32_k(float* p, float v, int n) {
  int t = blockIdx.x * blockDim.x + threadIdx.x;
  if (t < n) p[t] = v;
}

__global__ void cast_f32_bf16_k(const float* __restrict__ in,
                                unsigned short* __restrict__ out, int n) {
  int t = blockIdx.x * blockDim.x + threadIdx.x;
  if (t < n) out[t] = f2bf(in[t]);
}

// W[K, Nc] row-major (f32)  ->  Wt[Nc, K] row-major (bf16)
__global__ void transpose_cast_k(const float* __restrict__ W,
                                 unsigned short* __restrict__ Wt,
                                 int K, int Nc) {
  int t = blockIdx.x * blockDim.x + threadIdx.x;
  if (t >= K * Nc) return;
  int n = t / K, k = t - n * K;
  Wt[t] = f2bf(W[(size_t)k * Nc + n]);
}

// ---------------------------------------------------------------- WMMA GEMM
// C[M,Ncols] = A[M,K](bf16) * Bt[Ncols,K]^T(bf16) + bias, one wave per 16x16 tile.
template<bool OUT_BF16>
__global__ void gemm_bf16_wmma_k(const unsigned short* __restrict__ A,
                                 const unsigned short* __restrict__ Bt,
                                 const float* __restrict__ bias,
                                 void* __restrict__ Cout,
                                 int M, int K, int Ncols) {
  const int lane  = threadIdx.x;              // 0..31
  const int wave  = threadIdx.y;
  const int tileM = blockIdx.x;
  const int tileN = blockIdx.y * blockDim.y + wave;
  if (tileN * 16 >= Ncols) return;            // wave-uniform

  const int r  = lane & 15;
  const int hi = lane >> 4;                   // half-wave select
  // A layout (16-bit 16x32): lanes 0-15 hold K 0..15; lanes 16-31 K 16..31 (contig, matches B)
  const __bf16* arow = (const __bf16*)A + (size_t)(tileM * 16 + r) * K + hi * 16;
  // B layout (32x16): lane = column; lanes 0-15 K 0..15, lanes 16-31 K 16..31 (contig 32B)
  const __bf16* brow = (const __bf16*)Bt + (size_t)(tileN * 16 + r) * K + hi * 16;

  v8f acc = {};
  for (int k = 0; k < K; k += 32) {
    v8bf a0 = *(const v8bf*)(arow + k);
    v8bf a1 = *(const v8bf*)(arow + k + 8);
    v16bf a = __builtin_shufflevector(a0, a1, 0,1,2,3,4,5,6,7,8,9,10,11,12,13,14,15);
    v16bf b = *(const v16bf*)(brow + k);
    acc = __builtin_amdgcn_wmma_f32_16x16x32_bf16(false, a, false, b,
                                                  (short)0, acc, false, false);
  }

  // C layout: lane -> column, VGPR j -> row j + 8*hi
  const int col = tileN * 16 + r;
  const float bv = bias[col];
  const int rowBase = tileM * 16 + hi * 8;
#pragma unroll
  for (int j = 0; j < 8; ++j) {
    float v = acc[j] + bv;
    size_t idx = (size_t)(rowBase + j) * Ncols + col;
    if (OUT_BF16) ((unsigned short*)Cout)[idx] = f2bf(v);
    else          ((float*)Cout)[idx] = v;
  }
}

// ---------------------------------------------------------------- edge phase
__global__ void edge_scores_k(const long long* __restrict__ ei,
                              const unsigned short* __restrict__ q,
                              const unsigned short* __restrict__ k,
                              float* __restrict__ scores,
                              float* __restrict__ m) {
  int t = blockIdx.x * blockDim.x + threadIdx.x;
  if (t >= NE * NH) return;
  int e = t >> 3, h = t & 7;
  int src = (int)ei[e];
  int dst = (int)ei[NE + e];
  const v8bf* qp = (const v8bf*)(q + (size_t)dst * HHID + h * HID);
  const v8bf* kp = (const v8bf*)(k + (size_t)src * HHID + h * HID);
  float s = 0.f;
#pragma unroll
  for (int c = 0; c < 8; ++c) {
    v8bf qa = qp[c], ka = kp[c];
#pragma unroll
    for (int i = 0; i < 8; ++i) s += (float)qa[i] * (float)ka[i];
  }
  s *= 0.125f;                                 // 1/sqrt(HID)
  scores[t] = s;
  float* maddr = &m[(size_t)dst * NH + h];     // float max via int/uint atomics
  if (s >= 0.f) atomicMax((int*)maddr, __float_as_int(s));
  else          atomicMin((unsigned int*)maddr, __float_as_uint(s));
}

__global__ void edge_exp_k(const long long* __restrict__ ei,
                           float* __restrict__ scores,
                           const float* __restrict__ m,
                           float* __restrict__ den) {
  int t = blockIdx.x * blockDim.x + threadIdx.x;
  if (t >= NE * NH) return;
  int e = t >> 3, h = t & 7;
  int dst = (int)ei[NE + e];
  float ex = __expf(scores[t] - m[(size_t)dst * NH + h]);
  scores[t] = ex;
  atomicAdd(&den[(size_t)dst * NH + h], ex);
}

__global__ void edge_agg_k(const long long* __restrict__ ei,
                           const float* __restrict__ scores,
                           const float* __restrict__ den,
                           const unsigned short* __restrict__ v,
                           float* __restrict__ agg) {
  int t = blockIdx.x * blockDim.x + threadIdx.x;
  if (t >= NE * NH) return;
  int e = t >> 3, h = t & 7;
  int src = (int)ei[e];
  int dst = (int)ei[NE + e];
  float w = scores[t] / (den[(size_t)dst * NH + h] + 1e-16f);
  const v8bf* vp = (const v8bf*)(v + (size_t)src * HHID + h * HID);
  float* ap = agg + (size_t)dst * HHID + h * HID;
#pragma unroll
  for (int c = 0; c < 8; ++c) {
    v8bf va = vp[c];
#pragma unroll
    for (int i = 0; i < 8; ++i) atomicAdd(&ap[c * 8 + i], w * (float)va[i]);
  }
}

// head-mean + skip + relu ; write f32 (for pooling) and bf16 (next layer A)
__global__ void node_finish_k(const float* __restrict__ agg,
                              const float* __restrict__ skip,
                              float* __restrict__ hout,
                              unsigned short* __restrict__ xbout) {
  int t = blockIdx.x * blockDim.x + threadIdx.x;
  if (t >= NN * HID) return;
  int node = t >> 6, d = t & 63;
  float s = 0.f;
#pragma unroll
  for (int h = 0; h < NH; ++h) s += agg[(size_t)node * HHID + h * HID + d];
  float v = s * (1.0f / NH) + skip[t];
  v = fmaxf(v, 0.f);
  hout[t]  = v;
  xbout[t] = f2bf(v);
}

// ---------------------------------------------------------------- pooling
__global__ void pool_sum_k(const long long* __restrict__ batch,
                           const float* __restrict__ h,
                           float* __restrict__ sums) {
  int t = blockIdx.x * blockDim.x + threadIdx.x;
  if (t >= NN * HID) return;
  int node = t >> 6, d = t & 63;
  atomicAdd(&sums[(size_t)((int)batch[node]) * HID + d], h[t]);
}
__global__ void pool_cnt_k(const long long* __restrict__ batch,
                           float* __restrict__ cnt) {
  int n = blockIdx.x * blockDim.x + threadIdx.x;
  if (n >= NN) return;
  atomicAdd(&cnt[(int)batch[n]], 1.0f);
}

// ---------------------------------------------------------------- MLP + DynAAM head (tiny)
__global__ void head_k(const float* __restrict__ sums, const float* __restrict__ cnt,
                       const float* __restrict__ fc1W, const float* __restrict__ fc1b,
                       const float* __restrict__ fc2W, const float* __restrict__ fc2b,
                       const float* __restrict__ Wc, float* __restrict__ out) {
  __shared__ float g[HID], tb[HID], red[HID];
  int b = blockIdx.x, d = threadIdx.x;
  float c = fmaxf(cnt[b], 1.0f);
  g[d] = sums[b * HID + d] / c;
  __syncthreads();
  float acc = fc1b[d];
  for (int k = 0; k < HID; ++k) acc += g[k] * fc1W[k * HID + d];
  tb[d] = fmaxf(acc, 0.f);
  __syncthreads();
  acc = fc2b[d];
  for (int k = 0; k < HID; ++k) acc += tb[k] * fc2W[k * HID + d];
  float gv = fmaxf(acc, 0.f);
  __syncthreads();
  g[d] = gv;
  red[d] = gv * gv;
  __syncthreads();
  for (int s = HID / 2; s > 0; s >>= 1) {
    if (d < s) red[d] += red[d + s];
    __syncthreads();
  }
  float nrm = sqrtf(red[0]) + 1e-12f;
  tb[d] = g[d] / nrm;
  __syncthreads();
  if (d < NCL) {
    float dot = 0.f, wn = 0.f;
    for (int k = 0; k < HID; ++k) {
      float w = Wc[d * HID + k];
      dot += tb[k] * w;
      wn  += w * w;
    }
    out[b * NCL + d] = 30.0f * dot / (sqrtf(wn) + 1e-12f);
  }
}

// ---------------------------------------------------------------- driver
extern "C" void kernel_launch(void* const* d_in, const int* in_sizes, int n_in,
                              void* d_out, int out_size, void* d_ws, size_t ws_size,
                              hipStream_t stream) {
  (void)in_sizes; (void)n_in; (void)out_size; (void)ws_size;
  const float*     x     = (const float*)d_in[0];
  const long long* ei    = (const long long*)d_in[1];
  const long long* batch = (const long long*)d_in[2];
  const float* fc1W = (const float*)d_in[27];
  const float* fc1b = (const float*)d_in[28];
  const float* fc2W = (const float*)d_in[29];
  const float* fc2b = (const float*)d_in[30];
  const float* Wc   = (const float*)d_in[31];

  char* ws = (char*)d_ws;
  size_t off = 0;
  auto take = [&](size_t bytes) -> void* {
    void* p = ws + off;
    off += (bytes + 255) & ~(size_t)255;
    return p;
  };
  unsigned short* xb   = (unsigned short*)take((size_t)NN * DIN * 2);
  unsigned short* qb   = (unsigned short*)take((size_t)NN * HHID * 2);
  unsigned short* kb   = (unsigned short*)take((size_t)NN * HHID * 2);
  unsigned short* vb   = (unsigned short*)take((size_t)NN * HHID * 2);
  float* skip   = (float*)take((size_t)NN * HID * 4);
  float* scores = (float*)take((size_t)NE * NH * 4);
  float* mbuf   = (float*)take((size_t)NN * NH * 4);
  float* den    = (float*)take((size_t)NN * NH * 4);
  float* agg    = (float*)take((size_t)NN * HHID * 4);
  float* hcur   = (float*)take((size_t)NN * HID * 4);
  unsigned short* wtq = (unsigned short*)take((size_t)HHID * DIN * 2);
  unsigned short* wtk = (unsigned short*)take((size_t)HHID * DIN * 2);
  unsigned short* wtv = (unsigned short*)take((size_t)HHID * DIN * 2);
  unsigned short* wts = (unsigned short*)take((size_t)HID  * DIN * 2);
  float* sums = (float*)take((size_t)NGR * HID * 4);
  float* cnt  = (float*)take((size_t)NGR * 4);

  const int TPB = 256;
  const int EH  = NE * NH;

  for (int l = 0; l < 3; ++l) {
    const int din  = (l == 0) ? DIN : HID;
    const int base = 3 + l * 8;
    const float* pWq = (const float*)d_in[base + 0];
    const float* pbq = (const float*)d_in[base + 1];
    const float* pWk = (const float*)d_in[base + 2];
    const float* pbk = (const float*)d_in[base + 3];
    const float* pWv = (const float*)d_in[base + 4];
    const float* pbv = (const float*)d_in[base + 5];
    const float* pWs = (const float*)d_in[base + 6];
    const float* pbs = (const float*)d_in[base + 7];

    const int wqk = din * HHID, wss = din * HID;
    transpose_cast_k<<<(wqk + TPB - 1) / TPB, TPB, 0, stream>>>(pWq, wtq, din, HHID);
    transpose_cast_k<<<(wqk + TPB - 1) / TPB, TPB, 0, stream>>>(pWk, wtk, din, HHID);
    transpose_cast_k<<<(wqk + TPB - 1) / TPB, TPB, 0, stream>>>(pWv, wtv, din, HHID);
    transpose_cast_k<<<(wss + TPB - 1) / TPB, TPB, 0, stream>>>(pWs, wts, din, HID);
    if (l == 0)
      cast_f32_bf16_k<<<(NN * DIN + TPB - 1) / TPB, TPB, 0, stream>>>(x, xb, NN * DIN);

    dim3 gq(NN / 16, HHID / 16 / 8), bq(32, 8);           // 1250 x 4 blocks, 8 waves
    gemm_bf16_wmma_k<true><<<gq, bq, 0, stream>>>(xb, wtq, pbq, qb, NN, din, HHID);
    gemm_bf16_wmma_k<true><<<gq, bq, 0, stream>>>(xb, wtk, pbk, kb, NN, din, HHID);
    gemm_bf16_wmma_k<true><<<gq, bq, 0, stream>>>(xb, wtv, pbv, vb, NN, din, HHID);
    dim3 gs(NN / 16, 1), bs(32, HID / 16);                // skip: 4 N-tiles
    gemm_bf16_wmma_k<false><<<gs, bs, 0, stream>>>(xb, wts, pbs, skip, NN, din, HID);

    fill_f32_k<<<(NN * NH + TPB - 1) / TPB, TPB, 0, stream>>>(mbuf, -3.0e38f, NN * NH);
    fill_f32_k<<<(NN * NH + TPB - 1) / TPB, TPB, 0, stream>>>(den, 0.f, NN * NH);
    fill_f32_k<<<(NN * HHID + TPB - 1) / TPB, TPB, 0, stream>>>(agg, 0.f, NN * HHID);

    edge_scores_k<<<(EH + TPB - 1) / TPB, TPB, 0, stream>>>(ei, qb, kb, scores, mbuf);
    edge_exp_k<<<(EH + TPB - 1) / TPB, TPB, 0, stream>>>(ei, scores, mbuf, den);
    edge_agg_k<<<(EH + TPB - 1) / TPB, TPB, 0, stream>>>(ei, scores, den, vb, agg);
    node_finish_k<<<(NN * HID + TPB - 1) / TPB, TPB, 0, stream>>>(agg, skip, hcur, xb);
  }

  fill_f32_k<<<(NGR * HID + TPB - 1) / TPB, TPB, 0, stream>>>(sums, 0.f, NGR * HID);
  fill_f32_k<<<(NGR + TPB - 1) / TPB, TPB, 0, stream>>>(cnt, 0.f, NGR);
  pool_sum_k<<<(NN * HID + TPB - 1) / TPB, TPB, 0, stream>>>(batch, hcur, sums);
  pool_cnt_k<<<(NN + TPB - 1) / TPB, TPB, 0, stream>>>(batch, cnt);
  head_k<<<NGR, HID, 0, stream>>>(sums, cnt, fc1W, fc1b, fc2W, fc2b, Wc, (float*)d_out);
}
